// SimilarityMatrixFormation_53944789238457
// MI455X (gfx1250) — matrix operand order repaired and checked by reading, not compiled, not run on hardware
//
#include <hip/hip_runtime.h>

typedef __attribute__((ext_vector_type(8)))  float  v8f;
typedef __attribute__((ext_vector_type(8)))  __bf16 v8bf;
typedef __attribute__((ext_vector_type(16))) __bf16 v16bf;

#define B_DIM 64
#define T_DIM 2048
#define J_DIM 128
#define D_DIM 512
#define TILE_M 128
#define KC 32
#define LDSR 40   // LDS row stride in bf16 elems (32 data + 8 pad) -> conflict-free frags

__device__ __forceinline__ unsigned short f2bf(float x) {
  unsigned u = __float_as_uint(x);
  u += 0x7FFFu + ((u >> 16) & 1u);          // round-to-nearest-even
  return (unsigned short)(u >> 16);
}
__device__ __forceinline__ float bf2f(unsigned short h) {
  return __uint_as_float(((unsigned)h) << 16);
}

// Split 4 floats into bf16 hi/lo pairs and store 8B each to LDS.
__device__ __forceinline__ void split_store4(unsigned short* hi, unsigned short* lo,
                                             int off, float4 v) {
  unsigned short h0 = f2bf(v.x), h1 = f2bf(v.y), h2 = f2bf(v.z), h3 = f2bf(v.w);
  unsigned short l0 = f2bf(v.x - bf2f(h0));
  unsigned short l1 = f2bf(v.y - bf2f(h1));
  unsigned short l2 = f2bf(v.z - bf2f(h2));
  unsigned short l3 = f2bf(v.w - bf2f(h3));
  uint2 ph, pl;
  ph.x = (unsigned)h0 | ((unsigned)h1 << 16); ph.y = (unsigned)h2 | ((unsigned)h3 << 16);
  pl.x = (unsigned)l0 | ((unsigned)l1 << 16); pl.y = (unsigned)l2 | ((unsigned)l3 << 16);
  *(uint2*)(hi + off) = ph;
  *(uint2*)(lo + off) = pl;
}

// A fragment (16-bit 16x32): lane m<16 holds K {0..7}U{16..23}; lane m+16: {8..15}U{24..31}
__device__ __forceinline__ v16bf ld_fragA(const unsigned short* p, int off) {
  v8bf a = *(const v8bf*)(p + off);
  v8bf b = *(const v8bf*)(p + off + 16);
  return __builtin_shufflevector(a, b, 0,1,2,3,4,5,6,7,8,9,10,11,12,13,14,15);
}
// B fragment (16-bit 32x16): lane n<16 holds K {0..7}U{16..23} for col n; lane n+16: {8..15}U{24..31}
__device__ __forceinline__ v16bf ld_fragB(const unsigned short* p, int off) {
  v8bf a = *(const v8bf*)(p + off);
  v8bf b = *(const v8bf*)(p + off + 16);
  return __builtin_shufflevector(a, b, 0,1,2,3,4,5,6,7,8,9,10,11,12,13,14,15);
}

// Pre-kernel: qb[b*J + j] = bias + dot(U[b,j,:], w_q). One wave per (b,j).
__global__ __launch_bounds__(256)
void qry_bias_kernel(const float* __restrict__ U, const float* __restrict__ kern,
                     const float* __restrict__ bias, float* __restrict__ qb) {
  const int wave = (blockIdx.x * blockDim.x + threadIdx.x) >> 5;   // 0 .. B*J-1
  const int lane = threadIdx.x & 31;
  const float* row = U + (size_t)wave * D_DIM;
  const float* wq  = kern + D_DIM;          // kernel[512:1024] = w_q
  float s = 0.f;
  #pragma unroll
  for (int i = 0; i < 4; ++i) {
    int d = lane * 4 + i * 128;
    float4 u = *(const float4*)(row + d);
    float4 w = *(const float4*)(wq + d);
    s += u.x * w.x + u.y * w.y + u.z * w.z + u.w * w.w;
  }
  #pragma unroll
  for (int off = 16; off > 0; off >>= 1) s += __shfl_xor(s, off, 32);
  if (lane == 0) qb[wave] = s + bias[0];
}

// Main kernel: one 128x128 output tile per block, split-bf16 WMMA GEMM.
// S[t,j] = sum_d H[t,d] * (w_m[d]*U[j,d] + w_c[d])  +  (qry[j] + bias)
__global__ __launch_bounds__(256)
void sim_matrix_kernel(const float* __restrict__ H, const float* __restrict__ U,
                       const float* __restrict__ kern, const float* __restrict__ qb,
                       float* __restrict__ S) {
  __shared__ __align__(16) unsigned short sm[4 * TILE_M * LDSR];  // 40 KB
  unsigned short* sAh = sm;
  unsigned short* sAl = sm + TILE_M * LDSR;
  unsigned short* sBh = sm + 2 * TILE_M * LDSR;
  unsigned short* sBl = sm + 3 * TILE_M * LDSR;

  const int tid  = threadIdx.x;
  const int lane = tid & 31, wv = tid >> 5;
  const int half = lane >> 4, lm = lane & 15;
  const int b  = blockIdx.y;
  const int m0 = blockIdx.x * TILE_M;

  const float* Hb = H + ((size_t)b * T_DIM + m0) * D_DIM;
  const float* Ub = U + (size_t)b * J_DIM * D_DIM;

  v8f zero = {};
  v8f acc[8];
  #pragma unroll
  for (int i = 0; i < 8; ++i) acc[i] = zero;

  for (int k0 = 0; k0 < D_DIM; k0 += KC) {
    __syncthreads();   // previous compute done before overwriting LDS
    // Fill: 128 rows x 32 cols, fp32 -> (hi,lo) bf16. 1024 float4 / 256 thr = 4 each.
    #pragma unroll
    for (int i = 0; i < 4; ++i) {
      int idx = i * 256 + tid;
      int row = idx >> 3, c4 = idx & 7;
      int gk = k0 + c4 * 4;
      int soff = row * LDSR + c4 * 4;
      float4 h = *(const float4*)(Hb + (size_t)row * D_DIM + gk);
      split_store4(sAh, sAl, soff, h);
      float4 u  = *(const float4*)(Ub + (size_t)row * D_DIM + gk);
      float4 wm = *(const float4*)(kern + 2 * D_DIM + gk);   // w_m
      float4 wc = *(const float4*)(kern + gk);               // w_c (folds ctx term)
      float4 t;
      t.x = u.x * wm.x + wc.x; t.y = u.y * wm.y + wc.y;
      t.z = u.z * wm.z + wc.z; t.w = u.w * wm.w + wc.w;
      split_store4(sBh, sBl, soff, t);
    }
    __syncthreads();

    const int aoff = (wv * 16 + lm) * LDSR + half * 8;
    v16bf ah = ld_fragA(sAh, aoff);
    v16bf al = ld_fragA(sAl, aoff);
    #pragma unroll
    for (int nt = 0; nt < 8; ++nt) {
      const int boff = (nt * 16 + lm) * LDSR + half * 8;
      v16bf bh = ld_fragB(sBh, boff);
      v16bf bl = ld_fragB(sBl, boff);
      acc[nt] = __builtin_amdgcn_wmma_f32_16x16x32_bf16(false, ah, false, bh,
                                                        (short)0, acc[nt], false, false);
      acc[nt] = __builtin_amdgcn_wmma_f32_16x16x32_bf16(false, ah, false, bl,
                                                        (short)0, acc[nt], false, false);
      acc[nt] = __builtin_amdgcn_wmma_f32_16x16x32_bf16(false, al, false, bh,
                                                        (short)0, acc[nt], false, false);
    }
  }

  // Epilogue: C/D layout -> lane covers col lm; VGPR r covers row half*8 + r.
  const int rbase = m0 + wv * 16 + half * 8;
  #pragma unroll
  for (int nt = 0; nt < 8; ++nt) {
    const int col = nt * 16 + lm;
    const float q = qb[b * J_DIM + col];
    #pragma unroll
    for (int r = 0; r < 8; ++r) {
      S[((size_t)b * T_DIM + rbase + r) * J_DIM + col] = acc[nt][r] + q;
    }
  }
}

extern "C" void kernel_launch(void* const* d_in, const int* in_sizes, int n_in,
                              void* d_out, int out_size, void* d_ws, size_t ws_size,
                              hipStream_t stream) {
  (void)in_sizes; (void)n_in; (void)out_size; (void)ws_size;
  const float* H    = (const float*)d_in[0];
  const float* U    = (const float*)d_in[1];
  const float* kern = (const float*)d_in[2];
  const float* bias = (const float*)d_in[3];
  float* S  = (float*)d_out;
  float* qb = (float*)d_ws;                 // B*J floats = 32 KB scratch

  // 8192 (b,j) dots, 8 waves per block -> 1024 blocks
  qry_bias_kernel<<<dim3(B_DIM * J_DIM / 8), 256, 0, stream>>>(U, kern, bias, qb);
  sim_matrix_kernel<<<dim3(T_DIM / TILE_M, B_DIM), 256, 0, stream>>>(H, U, kern, qb, S);
}
